// BinaryLinear_30511447670910
// MI455X (gfx1250) — 8-bit matrix operand order repaired and checked by reading, compiled, not run on hardware
//
#include <hip/hip_runtime.h>
#include <hip/hip_bf16.h>


typedef __attribute__((ext_vector_type(16))) int   v16i;
typedef __attribute__((ext_vector_type(8)))  float v8f;

#define B_SZ   8192
#define IN_SZ  4096
#define OUT_SZ 4096
#define KTILES (IN_SZ / 128)
#define TILE_BYTES 2048
#define A_TILE_STRIDE (KTILES * TILE_BYTES)
#define B_TILE_STRIDE (KTILES * TILE_BYTES)

__device__ __forceinline__ unsigned sign_fp8(float f) {
  unsigned u = __float_as_uint(f);
  unsigned b = 0x38u | ((u >> 24) & 0x80u);
  return (f == 0.0f) ? 0u : b;
}

__global__ void pack_x_fp8(const float4* __restrict__ x, unsigned char* __restrict__ ap) {
  unsigned idx = blockIdx.x * blockDim.x + threadIdx.x;
  unsigned row = idx >> 10;
  unsigned k   = (idx & 1023u) << 2;
  float4 v = x[idx];
  unsigned d = sign_fp8(v.x) | (sign_fp8(v.y) << 8) |
               (sign_fp8(v.z) << 16) | (sign_fp8(v.w) << 24);
  unsigned r  = row & 15u, rowTile = row >> 4;
  unsigned kt = k >> 7,    kk = k & 127u;
  unsigned k64 = kk >> 6,  k0 = kk & 63u;
  unsigned hi   = (k0 >> 3) & 1u;
  unsigned lane = r | (hi << 4);
  unsigned vgpr = k64 * 8u + (k0 >> 4) * 2u + ((k0 & 7u) >> 2);
  unsigned off = (rowTile * KTILES + kt) * TILE_BYTES + lane * 64u + vgpr * 4u;
  *(unsigned*)(ap + off) = d;
}

__global__ void pack_w_fp8(const float4* __restrict__ w, unsigned char* __restrict__ bp) {
  unsigned idx = blockIdx.x * blockDim.x + threadIdx.x;
  unsigned row = idx >> 10;
  unsigned k   = (idx & 1023u) << 2;
  float4 v = w[idx];
  unsigned d = sign_fp8(v.x) | (sign_fp8(v.y) << 8) |
               (sign_fp8(v.z) << 16) | (sign_fp8(v.w) << 24);
  unsigned c  = row & 15u, colTile = row >> 4;
  unsigned kt = k >> 7,    kk = k & 127u;
  unsigned h    = (kk >> 3) & 1u;
  unsigned lane = c | (h << 4);
  unsigned vgpr = (kk >> 6) * 8u + ((kk >> 4) & 3u) * 2u + ((kk >> 2) & 1u);
  unsigned off = (colTile * KTILES + kt) * TILE_BYTES + lane * 64u + vgpr * 4u;
  *(unsigned*)(bp + off) = d;
}

__global__ __launch_bounds__(256, 1)
void bgemm_fp8_wmma(const unsigned char* __restrict__ ap,
                    const unsigned char* __restrict__ bp,
                    const float* __restrict__ bias,
                    float* __restrict__ out) {
  const int lane = threadIdx.x & 31;
  const int wave = threadIdx.x >> 5;
  const int wrow = wave >> 2;
  const int wcol = wave & 3;

  const int aTile0 = blockIdx.x * 8 + wrow * 4;
  const int bTile0 = blockIdx.y * 8 + wcol * 2;

  const unsigned char* aPtr = ap + (size_t)aTile0 * A_TILE_STRIDE + (unsigned)lane * 64u;
  const unsigned char* bPtr = bp + (size_t)bTile0 * B_TILE_STRIDE + (unsigned)lane * 64u;

  v8f acc[4][2];
#pragma unroll
  for (int i = 0; i < 4; ++i)
#pragma unroll
    for (int j = 0; j < 2; ++j)
      acc[i][j] = (v8f){0.f, 0.f, 0.f, 0.f, 0.f, 0.f, 0.f, 0.f};

  for (int kt = 0; kt < KTILES; ++kt) {
    const unsigned char* ak = aPtr + kt * TILE_BYTES;
    const unsigned char* bk = bPtr + kt * TILE_BYTES;

    __builtin_prefetch(ak + TILE_BYTES, 0, 3);
    __builtin_prefetch(bk + TILE_BYTES, 0, 3);

    v16i a0 = *(const v16i*)(ak + 0 * A_TILE_STRIDE);
    v16i a1 = *(const v16i*)(ak + 1 * A_TILE_STRIDE);
    v16i a2 = *(const v16i*)(ak + 2 * A_TILE_STRIDE);
    v16i a3 = *(const v16i*)(ak + 3 * A_TILE_STRIDE);
    v16i b0 = *(const v16i*)(bk + 0 * B_TILE_STRIDE);
    v16i b1 = *(const v16i*)(bk + 1 * B_TILE_STRIDE);

    acc[0][0] = __builtin_amdgcn_wmma_f32_16x16x128_fp8_fp8(a0, b0, (short)0, acc[0][0], false, false);
    acc[0][1] = __builtin_amdgcn_wmma_f32_16x16x128_fp8_fp8(a0, b1, (short)0, acc[0][1], false, false);
    acc[1][0] = __builtin_amdgcn_wmma_f32_16x16x128_fp8_fp8(a1, b0, (short)0, acc[1][0], false, false);
    acc[1][1] = __builtin_amdgcn_wmma_f32_16x16x128_fp8_fp8(a1, b1, (short)0, acc[1][1], false, false);
    acc[2][0] = __builtin_amdgcn_wmma_f32_16x16x128_fp8_fp8(a2, b0, (short)0, acc[2][0], false, false);
    acc[2][1] = __builtin_amdgcn_wmma_f32_16x16x128_fp8_fp8(a2, b1, (short)0, acc[2][1], false, false);
    acc[3][0] = __builtin_amdgcn_wmma_f32_16x16x128_fp8_fp8(a3, b0, (short)0, acc[3][0], false, false);
    acc[3][1] = __builtin_amdgcn_wmma_f32_16x16x128_fp8_fp8(a3, b1, (short)0, acc[3][1], false, false);
  }

  const int rowBase = blockIdx.x * 128 + wrow * 64;
  const int colBase = blockIdx.y * 128 + wcol * 32;
  const int n    = lane & 15;
  const int rsel = (lane >> 4) * 8;

#pragma unroll
  for (int j = 0; j < 2; ++j) {
    const int col = colBase + j * 16 + n;
    const float bv = bias[col];
#pragma unroll
    for (int i = 0; i < 4; ++i) {
#pragma unroll
      for (int e = 0; e < 8; ++e) {
        const int row = rowBase + i * 16 + rsel + e;
        out[(size_t)row * OUT_SZ + col] = acc[i][j][e] + bv;
      }
    }
  }
}

extern "C" void kernel_launch(void* const* d_in, const int* in_sizes, int n_in,
                              void* d_out, int out_size, void* d_ws, size_t ws_size,
                              hipStream_t stream) {
  const float* x    = (const float*)d_in[0];
  const float* w    = (const float*)d_in[1];
  const float* bias = (const float*)d_in[2];
  float* out = (float*)d_out;

  unsigned char* ap = (unsigned char*)d_ws;
  unsigned char* bp = ap + (size_t)B_SZ * IN_SZ;

  pack_x_fp8<<<(B_SZ * IN_SZ / 4) / 256, 256, 0, stream>>>((const float4*)x, ap);
  pack_w_fp8<<<(OUT_SZ * IN_SZ / 4) / 256, 256, 0, stream>>>((const float4*)w, bp);

  dim3 grid(B_SZ / 128, OUT_SZ / 128);
  bgemm_fp8_wmma<<<grid, 256, 0, stream>>>(ap, bp, bias, out);
}
